// GRUWithSkip_9191230013854
// MI455X (gfx1250) — matrix operand order repaired and checked by reading, not compiled, not run on hardware
//
#include <hip/hip_runtime.h>

// GRU-with-skip for MI455X (gfx1250, wave32, WMMA bf16).
//   1) Convert x and all weights to bf16 in workspace.
//   2) rx/zx/nx/skip input projections: bf16 WMMA GEMMs (HBM-bound), bf16 out.
//   3) Sequential scan: ONE persistent kernel, 64 blocks * 8 waves = 512 waves
//      = exactly the 512 (16x16) tiles of the (128 x 1024) hidden state.
//      3 fused GEMM tiles (Whr/Whz/Whn, K=1024 -> 96 wmma/step) + gate math,
//      atomic grid barrier between timesteps. Weights (6MB bf16) L2-resident.
//   4) Fused skip-add + LayerNorm -> bf16.
//   5) Output projection WMMA GEMM -> f32 d_out.
// K-loops use unroll(2): enough ILP to overlap loads with WMMA, small enough
// to avoid the round-1 spill/VGPR-MSB pathology. Gate math uses hardware
// v_rcp_f32/v_exp_f32 instead of IEEE division chains.

#define B_  128
#define T_  1024
#define I_  512
#define H_  1024
#define O_  512
#define BT_ (B_ * T_)

typedef __attribute__((ext_vector_type(16))) __bf16 v16bf;
typedef __attribute__((ext_vector_type(8)))  __bf16 v8bf;
typedef __attribute__((ext_vector_type(8)))  float  v8f;

union Frag16 { v16bf v; struct { v8bf lo, hi; } h; };

// A fragment (16x32 bf16, M x K), row-major source with leading dim ld.
// lanes 0-15: M=lane, v0..3 = K 0..7, v4..7 = K 16..23
// lanes 16-31: M=lane-16, v0..3 = K 8..15, v4..7 = K 24..31
__device__ __forceinline__ v16bf load_frag_a(const __bf16* base, int ld, int lane) {
  const int hl = lane >> 4, l = lane & 15;
  const __bf16* p = base + (size_t)l * ld + hl * 8;
  Frag16 u;
  u.h.lo = *(const v8bf*)(p);
  u.h.hi = *(const v8bf*)(p + 16);
  return u.v;
}

// B fragment (32x16 bf16, K x N) where B = W^T and W is row-major (N rows, K cols).
// lanes 0-15: N=lane, K=0..7 then 16..23; lanes 16-31: N=lane-16, K=8..15 then 24..31.
__device__ __forceinline__ v16bf load_frag_b(const __bf16* wbase, int ld, int lane) {
  const int hl = lane >> 4, l = lane & 15;
  const __bf16* p = wbase + (size_t)l * ld + hl * 8;
  Frag16 u;
  u.h.lo = *(const v8bf*)(p);
  u.h.hi = *(const v8bf*)(p + 16);
  return u.v;
}

// Fast gate math: hardware v_exp_f32 (via exp2) + v_rcp_f32, no IEEE divides.
#define LOG2E_ 1.4426950408889634f
__device__ __forceinline__ float fast_sigmoid(float x) {
  // 1/(1+exp(-x))
  const float e = __builtin_amdgcn_exp2f(-x * LOG2E_);
  return __builtin_amdgcn_rcpf(1.0f + e);
}
__device__ __forceinline__ float fast_tanh(float x) {
  // tanh(x) = 1 - 2/(exp(2x)+1)
  const float e = __builtin_amdgcn_exp2f(x * (2.0f * LOG2E_));
  return 1.0f - 2.0f * __builtin_amdgcn_rcpf(e + 1.0f);
}

// ---------------------------------------------------------------- conversions
__global__ __launch_bounds__(256) void f2bf_kernel(const float* __restrict__ src,
                                                   __bf16* __restrict__ dst, size_t n) {
  size_t i = (size_t)blockIdx.x * blockDim.x + threadIdx.x;
  const size_t stride = (size_t)gridDim.x * blockDim.x;
  for (; i < n; i += stride) dst[i] = (__bf16)src[i];
}

__global__ __launch_bounds__(256) void init_kernel(float* __restrict__ hq,
                                                   __bf16* __restrict__ hs0,
                                                   unsigned* __restrict__ counter, int n) {
  const int i = blockIdx.x * blockDim.x + threadIdx.x;
  if (i == 0) *counter = 0u;
  if (i < n) { hq[i] = 0.0f; hs0[i] = (__bf16)0.0f; }
}

// ------------------------------------------------------------------- GEMM
// OUT = A(bf16, MxK) @ W^T(bf16, W is NxK) + bias.  8 waves/block; each wave
// owns one 16-row M tile and four 16-col N tiles (reuses the A fragment 4x).
template <bool OUT_BF16>
__global__ __launch_bounds__(256)
void gemm_bias_kernel(const __bf16* __restrict__ A, const __bf16* __restrict__ W,
                      const float* __restrict__ bias, void* __restrict__ outp,
                      int M, int N, int K) {
  const int lane = threadIdx.x & 31;
  const int wave = threadIdx.x >> 5;
  const int gw   = blockIdx.x * 8 + wave;
  const int ngrp = N >> 6;                 // groups of 4 N-tiles
  const int mt   = gw / ngrp;
  const int ng   = gw % ngrp;
  if (mt * 16 >= M) return;
  const int m0 = mt << 4;
  const int n0 = ng << 6;

  const __bf16* Ap = A + (size_t)m0 * K;
  const __bf16* W0 = W + (size_t)(n0 +  0) * K;
  const __bf16* W1 = W + (size_t)(n0 + 16) * K;
  const __bf16* W2 = W + (size_t)(n0 + 32) * K;
  const __bf16* W3 = W + (size_t)(n0 + 48) * K;

  v8f acc0 = {}, acc1 = {}, acc2 = {}, acc3 = {};
#pragma unroll 2
  for (int k0 = 0; k0 < K; k0 += 32) {
    const v16bf a  = load_frag_a(Ap + k0, K, lane);
    const v16bf b0 = load_frag_b(W0 + k0, K, lane);
    const v16bf b1 = load_frag_b(W1 + k0, K, lane);
    const v16bf b2 = load_frag_b(W2 + k0, K, lane);
    const v16bf b3 = load_frag_b(W3 + k0, K, lane);
    acc0 = __builtin_amdgcn_wmma_f32_16x16x32_bf16(false, a, false, b0, (short)0, acc0, false, false);
    acc1 = __builtin_amdgcn_wmma_f32_16x16x32_bf16(false, a, false, b1, (short)0, acc1, false, false);
    acc2 = __builtin_amdgcn_wmma_f32_16x16x32_bf16(false, a, false, b2, (short)0, acc2, false, false);
    acc3 = __builtin_amdgcn_wmma_f32_16x16x32_bf16(false, a, false, b3, (short)0, acc3, false, false);
  }

  const int hl = lane >> 4, l = lane & 15;
  v8f accs[4] = {acc0, acc1, acc2, acc3};
#pragma unroll
  for (int j = 0; j < 4; ++j) {
    const int col = n0 + 16 * j + l;
    const float bv = bias[col];
#pragma unroll
    for (int r = 0; r < 8; ++r) {
      const int row = m0 + r + 8 * hl;
      const float v = accs[j][r] + bv;
      if (OUT_BF16) ((__bf16*)outp)[(size_t)row * N + col] = (__bf16)v;
      else          ((float*)outp)[(size_t)row * N + col]  = v;
    }
  }
}

// ------------------------------------------------------------- GRU scan
// Persistent kernel: 64 blocks x 256 thr = 512 waves = 512 output tiles of h.
__global__ __launch_bounds__(256)
void gru_scan_kernel(const __bf16* __restrict__ whr, const __bf16* __restrict__ whz,
                     const __bf16* __restrict__ whn,
                     const float* __restrict__ bhr, const float* __restrict__ bhz,
                     const float* __restrict__ bhn,
                     const __bf16* __restrict__ rx, const __bf16* __restrict__ zx,
                     const __bf16* __restrict__ nx,
                     __bf16* __restrict__ hs,   // (T+1, B, H); slot 0 pre-zeroed
                     float* __restrict__ hq,    // (B, H) running fp32 state
                     unsigned* __restrict__ counter) {
  const int lane = threadIdx.x & 31;
  const int wave = threadIdx.x >> 5;
  const int tile = blockIdx.x * 8 + wave;   // 0..511
  const int mt = tile >> 6;                 // 0..7   (batch tiles)
  const int nt = tile & 63;                 // 0..63  (hidden tiles)
  const int m0 = mt << 4, n0 = nt << 4;
  const int hl = lane >> 4, l = lane & 15;
  const int col = n0 + l;
  const float br = bhr[col], bz = bhz[col], bn = bhn[col];

  const __bf16* wr = whr + (size_t)n0 * H_;
  const __bf16* wz = whz + (size_t)n0 * H_;
  const __bf16* wn = whn + (size_t)n0 * H_;

#pragma unroll 1
  for (int t = 0; t < T_; ++t) {
    const __bf16* hA = hs + (size_t)t * (B_ * H_) + (size_t)m0 * H_;
    v8f aR = {}, aZ = {}, aN = {};
#pragma unroll 2
    for (int k0 = 0; k0 < H_; k0 += 32) {
      const v16bf a  = load_frag_a(hA + k0, H_, lane);
      const v16bf b0 = load_frag_b(wr + k0, H_, lane);
      const v16bf b1 = load_frag_b(wz + k0, H_, lane);
      const v16bf b2 = load_frag_b(wn + k0, H_, lane);
      aR = __builtin_amdgcn_wmma_f32_16x16x32_bf16(false, a, false, b0, (short)0, aR, false, false);
      aZ = __builtin_amdgcn_wmma_f32_16x16x32_bf16(false, a, false, b1, (short)0, aZ, false, false);
      aN = __builtin_amdgcn_wmma_f32_16x16x32_bf16(false, a, false, b2, (short)0, aN, false, false);
    }
#pragma unroll
    for (int r = 0; r < 8; ++r) {
      const int row = m0 + r + 8 * hl;                       // batch index
      const size_t xi = ((size_t)row * T_ + t) * H_ + col;   // (B,T,H)
      const float rg = fast_sigmoid((float)rx[xi] + aR[r] + br);
      const float zg = fast_sigmoid((float)zx[xi] + aZ[r] + bz);
      const float ngate = fast_tanh((float)nx[xi] + rg * (aN[r] + bn));
      const size_t hidx = (size_t)row * H_ + col;
      const float hp = hq[hidx];
      const float hnew = (1.0f - zg) * ngate + zg * hp;
      hq[hidx] = hnew;
      hs[(size_t)(t + 1) * (B_ * H_) + hidx] = (__bf16)hnew;
    }
    // ---- grid-wide barrier between timesteps ----
    __threadfence();
    __syncthreads();
    if (threadIdx.x == 0) {
      __hip_atomic_fetch_add(counter, 1u, __ATOMIC_ACQ_REL, __HIP_MEMORY_SCOPE_AGENT);
      const unsigned tgt = (unsigned)gridDim.x * (unsigned)(t + 1);
      while (__hip_atomic_load(counter, __ATOMIC_ACQUIRE, __HIP_MEMORY_SCOPE_AGENT) < tgt)
        __builtin_amdgcn_s_sleep(1);
    }
    __syncthreads();
    __threadfence();
  }
}

// ------------------------------------------------------- skip + LayerNorm
__global__ __launch_bounds__(256)
void ln_kernel(const __bf16* __restrict__ hs, const __bf16* __restrict__ skip,
               const float* __restrict__ gamma, const float* __restrict__ beta,
               __bf16* __restrict__ normed) {
  const int row = blockIdx.x;          // b*T + t
  const int b = row >> 10;             // T_ == 1024
  const int t = row & (T_ - 1);
  const __bf16* hp = hs + ((size_t)(t + 1) * B_ + b) * H_;
  const __bf16* sp = skip + (size_t)row * H_;
  const int i0 = threadIdx.x * 4;
  float c[4];
  float s = 0.0f, ss = 0.0f;
#pragma unroll
  for (int j = 0; j < 4; ++j) {
    c[j] = (float)hp[i0 + j] + (float)sp[i0 + j];
    s += c[j]; ss += c[j] * c[j];
  }
#pragma unroll
  for (int o = 16; o >= 1; o >>= 1) {
    s  += __shfl_xor(s,  o, 32);
    ss += __shfl_xor(ss, o, 32);
  }
  __shared__ float sb[8], ssb[8];
  if ((threadIdx.x & 31) == 0) { sb[threadIdx.x >> 5] = s; ssb[threadIdx.x >> 5] = ss; }
  __syncthreads();
  float S = 0.0f, SS = 0.0f;
#pragma unroll
  for (int j = 0; j < 8; ++j) { S += sb[j]; SS += ssb[j]; }
  const float mu   = S * (1.0f / H_);
  const float var  = SS * (1.0f / H_) - mu * mu;
  const float rinv = rsqrtf(var + 1e-5f);
#pragma unroll
  for (int j = 0; j < 4; ++j) {
    const float nv = (c[j] - mu) * rinv * gamma[i0 + j] + beta[i0 + j];
    normed[(size_t)row * H_ + i0 + j] = (__bf16)nv;
  }
}

// ------------------------------------------------------------------ launch
extern "C" void kernel_launch(void* const* d_in, const int* in_sizes, int n_in,
                              void* d_out, int out_size, void* d_ws, size_t ws_size,
                              hipStream_t stream) {
  (void)in_sizes; (void)n_in; (void)out_size; (void)ws_size;
  const float* x     = (const float*)d_in[0];
  const float* Wir   = (const float*)d_in[1];
  const float* bir   = (const float*)d_in[2];
  const float* Whr   = (const float*)d_in[3];
  const float* bhr   = (const float*)d_in[4];
  const float* Wiz   = (const float*)d_in[5];
  const float* biz   = (const float*)d_in[6];
  const float* Whz   = (const float*)d_in[7];
  const float* bhz   = (const float*)d_in[8];
  const float* Win   = (const float*)d_in[9];
  const float* bin_  = (const float*)d_in[10];
  const float* Whn   = (const float*)d_in[11];
  const float* bhn   = (const float*)d_in[12];
  const float* Wskip = (const float*)d_in[13];
  const float* bskip = (const float*)d_in[14];
  const float* gamma = (const float*)d_in[15];
  const float* beta  = (const float*)d_in[16];
  const float* Wout  = (const float*)d_in[17];
  const float* bout  = (const float*)d_in[18];

  char* ws = (char*)d_ws;
  size_t off = 0;
  auto alloc = [&](size_t bytes) -> char* {
    char* p = ws + off;
    off += (bytes + 1023) & ~(size_t)1023;
    return p;
  };
  __bf16* xb     = (__bf16*)alloc((size_t)BT_ * I_ * 2);
  __bf16* wirb   = (__bf16*)alloc((size_t)H_ * I_ * 2);
  __bf16* wizb   = (__bf16*)alloc((size_t)H_ * I_ * 2);
  __bf16* winb   = (__bf16*)alloc((size_t)H_ * I_ * 2);
  __bf16* wskipb = (__bf16*)alloc((size_t)H_ * I_ * 2);
  __bf16* whrb   = (__bf16*)alloc((size_t)H_ * H_ * 2);
  __bf16* whzb   = (__bf16*)alloc((size_t)H_ * H_ * 2);
  __bf16* whnb   = (__bf16*)alloc((size_t)H_ * H_ * 2);
  __bf16* woutb  = (__bf16*)alloc((size_t)O_ * H_ * 2);
  __bf16* rx     = (__bf16*)alloc((size_t)BT_ * H_ * 2);
  __bf16* zx     = (__bf16*)alloc((size_t)BT_ * H_ * 2);
  __bf16* nx     = (__bf16*)alloc((size_t)BT_ * H_ * 2);
  __bf16* skip   = (__bf16*)alloc((size_t)BT_ * H_ * 2);
  __bf16* hs     = (__bf16*)alloc((size_t)(T_ + 1) * B_ * H_ * 2);
  __bf16* normed = (__bf16*)alloc((size_t)BT_ * H_ * 2);
  float*  hq     = (float*)alloc((size_t)B_ * H_ * 4);
  unsigned* counter = (unsigned*)alloc(1024);

  // 1) dtype conversions
  f2bf_kernel<<<4096, 256, 0, stream>>>(x, xb, (size_t)BT_ * I_);
  f2bf_kernel<<<512, 256, 0, stream>>>(Wir, wirb, (size_t)H_ * I_);
  f2bf_kernel<<<512, 256, 0, stream>>>(Wiz, wizb, (size_t)H_ * I_);
  f2bf_kernel<<<512, 256, 0, stream>>>(Win, winb, (size_t)H_ * I_);
  f2bf_kernel<<<512, 256, 0, stream>>>(Wskip, wskipb, (size_t)H_ * I_);
  f2bf_kernel<<<1024, 256, 0, stream>>>(Whr, whrb, (size_t)H_ * H_);
  f2bf_kernel<<<1024, 256, 0, stream>>>(Whz, whzb, (size_t)H_ * H_);
  f2bf_kernel<<<1024, 256, 0, stream>>>(Whn, whnb, (size_t)H_ * H_);
  f2bf_kernel<<<512, 256, 0, stream>>>(Wout, woutb, (size_t)O_ * H_);

  // 2) input-side projections: (BT x I) @ W^T -> bf16 (B,T,H)
  const int blocksIn = (BT_ / 16) * (H_ / 64) / 8;   // 16384
  gemm_bias_kernel<true><<<blocksIn, 256, 0, stream>>>(xb, wirb, bir, rx, BT_, H_, I_);
  gemm_bias_kernel<true><<<blocksIn, 256, 0, stream>>>(xb, wizb, biz, zx, BT_, H_, I_);
  gemm_bias_kernel<true><<<blocksIn, 256, 0, stream>>>(xb, winb, bin_, nx, BT_, H_, I_);
  gemm_bias_kernel<true><<<blocksIn, 256, 0, stream>>>(xb, wskipb, bskip, skip, BT_, H_, I_);

  // 3) recurrent scan (persistent, grid barrier per step)
  init_kernel<<<(B_ * H_ + 255) / 256, 256, 0, stream>>>(hq, hs, counter, B_ * H_);
  gru_scan_kernel<<<64, 256, 0, stream>>>(whrb, whzb, whnb, bhr, bhz, bhn,
                                          rx, zx, nx, hs, hq, counter);

  // 4) skip + LayerNorm
  ln_kernel<<<BT_, 256, 0, stream>>>(hs, skip, gamma, beta, normed);

  // 5) output projection: (BT x H) @ Wout^T -> f32 (B,T,O)
  const int blocksOut = (BT_ / 16) * (O_ / 64) / 8;  // 8192
  gemm_bias_kernel<false><<<blocksOut, 256, 0, stream>>>(normed, woutb, bout,
                                                         d_out, BT_, O_, H_);
}
